// NodeUpdate_47218870453042
// MI455X (gfx1250) — compile-verified, matrix operand order checked by reading, not run on hardware
//
#include <hip/hip_runtime.h>
#include <hip/hip_bf16.h>

#define DEV_INLINE __device__ __forceinline__

typedef __attribute__((ext_vector_type(16))) __bf16 v16bf;
typedef __attribute__((ext_vector_type(8)))  float  v8f;
typedef __attribute__((ext_vector_type(4)))  unsigned int u32x4;

static constexpr int N_NODES = 20000;
static constexpr int N_EDGES = 320000;
static constexpr int F       = 256;
static constexpr int H       = 1024;
static constexpr int OUTF    = 256;
static constexpr int K0      = 2 * F;                       // 512
static constexpr int MPAD    = ((N_NODES + 63) / 64) * 64;  // 20032
static constexpr float LN_EPS = 1e-5f;

// ---------- helpers ----------
DEV_INLINE unsigned short f2bf(float f) {
  unsigned int u = __float_as_uint(f);
  u += 0x7FFFu + ((u >> 16) & 1u);   // round-to-nearest-even
  return (unsigned short)(u >> 16);
}

// ---------- 1) zero agg ----------
__global__ void zero_f32(float* __restrict__ p, int n4) {
  int t = blockIdx.x * blockDim.x + threadIdx.x;
  if (t < n4) ((float4*)p)[t] = make_float4(0.f, 0.f, 0.f, 0.f);
}

// ---------- 2) scatter-add edge_attr into agg[col] ----------
__global__ void scatter_add(const float* __restrict__ ea,
                            const int* __restrict__ col,  // edge_index row 1
                            float* __restrict__ agg) {
  int t = blockIdx.x * blockDim.x + threadIdx.x;     // E * (F/4) threads
  if (t >= N_EDGES * (F / 4)) return;
  int e = t >> 6;                 // F/4 == 64
  int j = (t & 63) * 4;
  float4 v = ((const float4*)(ea + (long)e * F))[j >> 2];
  float* dst = agg + (long)col[e] * F + j;
  atomicAdd(dst + 0, v.x);
  atomicAdd(dst + 1, v.y);
  atomicAdd(dst + 2, v.z);
  atomicAdd(dst + 3, v.w);
}

// ---------- 3) h0 = bf16(concat(x, agg)), zero-padded to MPAD rows ----------
__global__ void build_h0(const float* __restrict__ x,
                         const float* __restrict__ agg,
                         unsigned short* __restrict__ h0) {
  int t = blockIdx.x * blockDim.x + threadIdx.x;     // MPAD * 512 threads
  if (t >= MPAD * K0) return;
  int row = t >> 9;               // /512
  int c   = t & 511;
  float v = 0.f;
  if (row < N_NODES)
    v = (c < F) ? x[(long)row * F + c] : agg[(long)row * F + (c - F)];
  h0[t] = f2bf(v);
}

// ---------- 4) Wt[n][k] = bf16(W[k][n]) ----------
__global__ void transpose_w(const float* __restrict__ W,
                            unsigned short* __restrict__ Wt,
                            int K, int Nout) {
  int t = blockIdx.x * blockDim.x + threadIdx.x;
  if (t >= K * Nout) return;
  int n = t / K, k = t - n * K;
  Wt[(long)n * K + k] = f2bf(W[(long)k * Nout + n]);
}

// ---------- 5) WMMA GEMM:  C[MPAD x Nout] = act(A[MPAD x K] @ Wt^T + b) ----------
// Block: 256 threads (8 waves), tile 64(M) x 128(N); wave grid 2(M) x 4(N),
// each wave computes 2x2 tiles of 16x16; K step = 64 -> 8 v_wmma per stage.
// A tile staged global->LDS with GLOBAL_LOAD_ASYNC_TO_LDS_B128 (ASYNCcnt),
// double-buffered so the async copy of stage k+64 overlaps the WMMAs of stage k.
union Frag { u32x4 u[2]; v16bf v; };

template <bool RELU, bool OUT_F32>
__global__ __launch_bounds__(256)
void gemm_wmma(const unsigned short* __restrict__ A,   // bf16 [MPAD x K]
               const unsigned short* __restrict__ Wt,  // bf16 [Nout x K]
               const float* __restrict__ bias,         // f32  [Nout]
               void* __restrict__ Cout,                // bf16 or f32 [MPAD x Nout]
               int K, int Nout) {
  constexpr int LDS_STRIDE = 72;       // halves; 144B rows: 16B-aligned, conflict-free
  constexpr int BUF = 64 * LDS_STRIDE; // halves per stage buffer
  __shared__ unsigned short ldsA[2 * BUF];

  const int lane = threadIdx.x & 31;
  const int wave = threadIdx.x >> 5;
  const int wm = wave & 1;        // M half (0..1) -> 32 rows each
  const int wn = wave >> 1;       // N quarter (0..3) -> 32 cols each
  const int blockM = blockIdx.x * 64;
  const int ncb    = blockIdx.y * 128 + wn * 32;
  const int mrow = lane & 15;
  const int kb   = (lane & 16) ? 8 : 0;   // per-lane K sub-base (ISA A/B layout)

  // A-tile staging: thread -> (row 0..63, two 16B chunks 64B apart)
  const int sr = threadIdx.x >> 2;          // row 0..63
  const int sc = (threadIdx.x & 3) * 8;     // halves 0/8/16/24 (first 64B of row)
  const unsigned short* aSrc = A + (long)(blockM + sr) * K + sc;
  // Low 32 bits of the generic shared pointer == wave-relative LDS byte offset
  const unsigned ldsOff0 = (unsigned)(unsigned long long)&ldsA[sr * LDS_STRIDE + sc];

  // Issue the async stage of A[blockM:blockM+64, k:k+64) into buffer `buf`.
  // INST_OFFSET applies to both LDS dst and global src (ISA 15.18.3 async):
  // chunk0 at +0, chunk1 at +64B -> halves [sc, sc+8) and [sc+32, sc+40).
  auto stage = [&](int k, int buf) {
    const unsigned short* g = aSrc + k;
    const unsigned off = ldsOff0 + (unsigned)buf * (BUF * 2);
    asm volatile("global_load_async_to_lds_b128 %0, %1, off\n\t"
                 "global_load_async_to_lds_b128 %0, %1, off offset:64"
                 :: "v"(off), "v"(g) : "memory");
  };

  v8f acc[2][2] = {};

  // Pipeline prologue: stage k=0, wait, publish.
  stage(0, 0);
  asm volatile("s_wait_asynccnt 0" ::: "memory");
  __syncthreads();

  int cur = 0;
  for (int k = 0; k < K; k += 64) {
    const bool more = (k + 64 < K);
    if (more) {
      stage(k + 64, cur ^ 1);   // overlaps with the WMMAs below
      __builtin_prefetch(Wt + (long)(ncb + mrow) * K + k + 64 + kb, 0, 3);
      __builtin_prefetch(Wt + (long)(ncb + 16 + mrow) * K + k + 64 + kb, 0, 3);
    }

#pragma unroll
    for (int ks = 0; ks < 64; ks += 32) {
      Frag bF[2];
#pragma unroll
      for (int nt = 0; nt < 2; ++nt) {
        const unsigned short* bp = Wt + (long)(ncb + nt * 16 + mrow) * K + k + ks + kb;
        bF[nt].u[0] = *(const u32x4*)bp;
        bF[nt].u[1] = *(const u32x4*)(bp + 16);
      }
      Frag aF[2];
#pragma unroll
      for (int mt = 0; mt < 2; ++mt) {
        const unsigned short* ap =
            &ldsA[cur * BUF + (wm * 32 + mt * 16 + mrow) * LDS_STRIDE + ks + kb];
        aF[mt].u[0] = *(const u32x4*)ap;
        aF[mt].u[1] = *(const u32x4*)(ap + 16);
      }
#pragma unroll
      for (int mt = 0; mt < 2; ++mt)
#pragma unroll
        for (int nt = 0; nt < 2; ++nt)
          acc[mt][nt] = __builtin_amdgcn_wmma_f32_16x16x32_bf16(
              false, aF[mt].v, false, bF[nt].v, (short)0, acc[mt][nt], false, false);
    }

    if (more) {
      asm volatile("s_wait_asynccnt 0" ::: "memory");  // next stage landed in LDS
      __syncthreads();  // all waves: done consuming cur, next stage published
    }
    cur ^= 1;
  }

  // Epilogue. C/D layout: VGPR r -> M = r (lanes 0-15) or r+8 (lanes 16-31), N = lane&15.
  const int chi = (lane & 16) ? 8 : 0;
#pragma unroll
  for (int mt = 0; mt < 2; ++mt) {
#pragma unroll
    for (int nt = 0; nt < 2; ++nt) {
      const int colg = ncb + nt * 16 + mrow;
      const float bv = bias[colg];
#pragma unroll
      for (int r = 0; r < 8; ++r) {
        const int rowg = blockM + wm * 32 + mt * 16 + chi + r;
        float v = acc[mt][nt][r] + bv;
        if (RELU) v = fmaxf(v, 0.f);
        if (OUT_F32)
          ((float*)Cout)[(long)rowg * Nout + colg] = v;
        else
          ((unsigned short*)Cout)[(long)rowg * Nout + colg] = f2bf(v);
      }
    }
  }
}

// ---------- 6) LayerNorm over OUT=256, one wave32 per row ----------
__global__ __launch_bounds__(256)
void layernorm(const float* __restrict__ h3, const float* __restrict__ gamma,
               const float* __restrict__ beta, float* __restrict__ out, int nrows) {
  const int lane = threadIdx.x & 31;
  const int row  = blockIdx.x * 8 + (threadIdx.x >> 5);
  if (row >= nrows) return;
  const float4* p = (const float4*)(h3 + (long)row * OUTF + lane * 8);
  float4 a = p[0], b = p[1];
  float vals[8] = {a.x, a.y, a.z, a.w, b.x, b.y, b.z, b.w};
  float s = 0.f, sq = 0.f;
#pragma unroll
  for (int j = 0; j < 8; ++j) { s += vals[j]; sq += vals[j] * vals[j]; }
#pragma unroll
  for (int off = 16; off >= 1; off >>= 1) {
    s  += __shfl_xor(s, off, 32);
    sq += __shfl_xor(sq, off, 32);
  }
  const float mean = s * (1.f / OUTF);
  const float var  = sq * (1.f / OUTF) - mean * mean;
  const float rs   = rsqrtf(var + LN_EPS);
#pragma unroll
  for (int j = 0; j < 8; ++j) {
    const int c = lane * 8 + j;
    out[(long)row * OUTF + c] = (vals[j] - mean) * rs * gamma[c] + beta[c];
  }
}

// ---------- launch ----------
extern "C" void kernel_launch(void* const* d_in, const int* in_sizes, int n_in,
                              void* d_out, int out_size, void* d_ws, size_t ws_size,
                              hipStream_t stream) {
  const float* x   = (const float*)d_in[0];
  const int*   ei  = (const int*)d_in[1];    // [2, E]; row 1 = destination
  const float* ea  = (const float*)d_in[2];
  // d_in[3]=u, d_in[4]=batch unused by the reference math
  const float* W0 = (const float*)d_in[5];  const float* b0 = (const float*)d_in[6];
  const float* W1 = (const float*)d_in[7];  const float* b1 = (const float*)d_in[8];
  const float* W2 = (const float*)d_in[9];  const float* b2 = (const float*)d_in[10];
  const float* W3 = (const float*)d_in[11]; const float* b3 = (const float*)d_in[12];
  const float* gamma = (const float*)d_in[13];
  const float* beta  = (const float*)d_in[14];
  float* out = (float*)d_out;

  // workspace carve-up (256B aligned)
  auto align256 = [](size_t v) { return (v + 255) & ~(size_t)255; };
  char* base = (char*)d_ws;
  size_t off = 0;
  float*          agg = (float*)(base + off);          off = align256(off + (size_t)N_NODES * F * 4);
  unsigned short* h0  = (unsigned short*)(base + off); off = align256(off + (size_t)MPAD * K0 * 2);
  unsigned short* hA  = (unsigned short*)(base + off); off = align256(off + (size_t)MPAD * H * 2);
  unsigned short* hB  = (unsigned short*)(base + off); off = align256(off + (size_t)MPAD * H * 2);
  float*          h3  = (float*)(base + off);          off = align256(off + (size_t)MPAD * OUTF * 4);
  unsigned short* W0t = (unsigned short*)(base + off); off = align256(off + (size_t)H * K0 * 2);
  unsigned short* W1t = (unsigned short*)(base + off); off = align256(off + (size_t)H * H * 2);
  unsigned short* W2t = (unsigned short*)(base + off); off = align256(off + (size_t)H * H * 2);
  unsigned short* W3t = (unsigned short*)(base + off); off = align256(off + (size_t)OUTF * H * 2);
  (void)ws_size; (void)n_in; (void)in_sizes; (void)out_size;

  const int B = 256;

  // 1) agg = 0
  { int n4 = N_NODES * F / 4;
    zero_f32<<<(n4 + B - 1) / B, B, 0, stream>>>(agg, n4); }

  // 2) scatter-add (col = edge_index row 1)
  { int nt = N_EDGES * (F / 4);
    scatter_add<<<(nt + B - 1) / B, B, 0, stream>>>(ea, ei + N_EDGES, agg); }

  // 3) weight transpose+convert
  transpose_w<<<(K0 * H + B - 1) / B, B, 0, stream>>>(W0, W0t, K0, H);
  transpose_w<<<(H * H + B - 1) / B, B, 0, stream>>>(W1, W1t, H, H);
  transpose_w<<<(H * H + B - 1) / B, B, 0, stream>>>(W2, W2t, H, H);
  transpose_w<<<(H * OUTF + B - 1) / B, B, 0, stream>>>(W3, W3t, H, OUTF);

  // 4) h0 = bf16([x | agg]) padded
  { long nt = (long)MPAD * K0;
    build_h0<<<(int)((nt + B - 1) / B), B, 0, stream>>>(x, agg, h0); }

  // 5) MLP
  dim3 blk(B);
  gemm_wmma<true,  false><<<dim3(MPAD / 64, H / 128),    blk, 0, stream>>>(h0, W0t, b0, hA, K0, H);
  gemm_wmma<true,  false><<<dim3(MPAD / 64, H / 128),    blk, 0, stream>>>(hA, W1t, b1, hB, H,  H);
  gemm_wmma<true,  false><<<dim3(MPAD / 64, H / 128),    blk, 0, stream>>>(hB, W2t, b2, hA, H,  H);
  gemm_wmma<false, true ><<<dim3(MPAD / 64, OUTF / 128), blk, 0, stream>>>(hA, W3t, b3, h3, H, OUTF);

  // 6) LayerNorm -> out
  layernorm<<<(N_NODES + 7) / 8, B, 0, stream>>>(h3, gamma, beta, out, N_NODES);
}
